// Critic_16681652977907
// MI455X (gfx1250) — matrix operand order repaired and checked by reading, not compiled, not run on hardware
//
#include <hip/hip_runtime.h>

typedef _Float16 v16h __attribute__((ext_vector_type(16)));
typedef _Float16 v8h  __attribute__((ext_vector_type(8)));
typedef _Float16 v4h  __attribute__((ext_vector_type(4)));
typedef float    v8f  __attribute__((ext_vector_type(8)));

#define DEVFN static __device__ __forceinline__

// Problem dims
constexpr int S_    = 128;
constexpr int ADIM_ = 16;
constexpr int H_    = 64;
constexpr int IN_   = S_ + ADIM_;        // 144
constexpr int NROWS = 65536 * 8;         // 524288
constexpr int XSTR  = 168;               // x row stride (halves), padded K=160 + bank pad
constexpr int HSTR  = 72;                // h row stride (halves), 64 + bank pad

// Weight layout in d_ws and low LDS (bytes)
constexpr int OFF_WENCT = 0;                          // f16 [64][168]  (W_enc^T, K-padded w/ zeros)
constexpr int OFF_WOBST = OFF_WENCT + 64 * XSTR * 2;  // f16 [64][72]
constexpr int OFF_WIT   = OFF_WOBST + 64 * HSTR * 2;  // f16 [192][72]
constexpr int OFF_WHT   = OFF_WIT   + 192 * HSTR * 2; // f16 [192][72]
constexpr int OFF_BENC  = OFF_WHT   + 192 * HSTR * 2; // f32 [64]
constexpr int OFF_BOBS  = OFF_BENC + 64 * 4;
constexpr int OFF_BI    = OFF_BOBS + 64 * 4;          // f32 [192]
constexpr int OFF_BH    = OFF_BI + 192 * 4;           // f32 [192]
constexpr int OFF_WDEC  = OFF_BH + 192 * 4;           // f32 [64]
constexpr int OFF_BDEC  = OFF_WDEC + 64 * 4;          // f32 [1] (padded to 16B)
constexpr int WBYTES    = OFF_BDEC + 16;              // 88336 bytes, 16B aligned
// LDS-only regions
constexpr int OFF_XBUF  = WBYTES;                     // f16 [128][168]
constexpr int OFF_HBUF  = OFF_XBUF + 128 * XSTR * 2;  // f16 [128][72]
constexpr int OFF_CBUF  = OFF_HBUF + 128 * HSTR * 2;  // f16 [128][72]
constexpr int SMEM_BYTES = OFF_CBUF + 128 * HSTR * 2; // 168208 bytes (< 320KB/WGP)

// ---------------- prep kernel: transpose + pad + f32->f16 weights into d_ws --------------
__global__ void prep_weights(const float* __restrict__ Wenc, const float* __restrict__ benc,
                             const float* __restrict__ Wobs, const float* __restrict__ bobs,
                             const float* __restrict__ Wi,   const float* __restrict__ bi,
                             const float* __restrict__ Wh,   const float* __restrict__ bh,
                             const float* __restrict__ Wdec, const float* __restrict__ bdec,
                             unsigned char* __restrict__ ws) {
  const int tid = blockIdx.x * blockDim.x + threadIdx.x;
  const int nth = gridDim.x * blockDim.x;
  _Float16* wencT = (_Float16*)(ws + OFF_WENCT);
  _Float16* wobsT = (_Float16*)(ws + OFF_WOBST);
  _Float16* wiT   = (_Float16*)(ws + OFF_WIT);
  _Float16* whT   = (_Float16*)(ws + OFF_WHT);
  for (int i = tid; i < 64 * XSTR; i += nth) {
    int n = i / XSTR, k = i % XSTR;
    wencT[i] = (_Float16)((k < IN_) ? Wenc[k * H_ + n] : 0.0f);
  }
  for (int i = tid; i < 64 * HSTR; i += nth) {
    int n = i / HSTR, k = i % HSTR;
    wobsT[i] = (_Float16)((k < H_) ? Wobs[k * H_ + n] : 0.0f);
  }
  for (int i = tid; i < 192 * HSTR; i += nth) {
    int n = i / HSTR, k = i % HSTR;
    wiT[i] = (_Float16)((k < H_) ? Wi[k * 192 + n] : 0.0f);
    whT[i] = (_Float16)((k < H_) ? Wh[k * 192 + n] : 0.0f);
  }
  for (int i = tid; i < 64; i += nth) {
    ((float*)(ws + OFF_BENC))[i] = benc[i];
    ((float*)(ws + OFF_BOBS))[i] = bobs[i];
    ((float*)(ws + OFF_WDEC))[i] = Wdec[i];
  }
  for (int i = tid; i < 192; i += nth) {
    ((float*)(ws + OFF_BI))[i] = bi[i];
    ((float*)(ws + OFF_BH))[i] = bh[i];
  }
  if (tid == 0) ((float*)(ws + OFF_BDEC))[0] = bdec[0];
}

// ---------------- WMMA fragment helpers (ISA 7.12.2 16-bit layouts) ----------------------
DEVFN v16h ldsA(const _Float16* base, int stride, int k0) {
  // A 16x32 f16: lane L<16 row L, halves[0..7]=K k0..k0+7, halves[8..15]=K k0+16..k0+23;
  // lanes 16..31: same rows, K offset +8.
  const int lane = threadIdx.x & 31;
  const _Float16* q = base + (lane & 15) * stride + k0 + ((lane & 16) ? 8 : 0);
  v8h lo = *(const v8h*)q;
  v8h hi = *(const v8h*)(q + 16);
  return __builtin_shufflevector(lo, hi, 0,1,2,3,4,5,6,7,8,9,10,11,12,13,14,15);
}
DEVFN v16h ldsB(const _Float16* base, int stride, int k0) {
  // B 32x16 f16 from W^T row-major: lane L = column n (L&15); halves[0..7]=K k0..k0+7,
  // halves[8..15]=K k0+16..k0+23; lanes>=16: K offset +8 (matches ldsA).
  const int lane = threadIdx.x & 31;
  const _Float16* q = base + (lane & 15) * stride + k0 + ((lane & 16) ? 8 : 0);
  v8h lo = *(const v8h*)q;
  v8h hi = *(const v8h*)(q + 16);
  return __builtin_shufflevector(lo, hi, 0,1,2,3,4,5,6,7,8,9,10,11,12,13,14,15);
}
DEVFN v8f wmma_f(v16h a, v16h b, v8f c) {
  return __builtin_amdgcn_wmma_f32_16x16x32_f16(false, a, false, b, (short)0, c, false, false);
}
DEVFN void storeC16(_Float16* buf, int stride, int t, v8f acc) {
  // C layout: VGPR r = row r (lanes 0-15) / row r+8 (lanes 16-31); col = 16t + lane%16
  const int lane = threadIdx.x & 31;
  const int col = 16 * t + (lane & 15);
  const int rb = (lane & 16) ? 8 : 0;
#pragma unroll
  for (int r = 0; r < 8; ++r) buf[(rb + r) * stride + col] = (_Float16)acc[r];
}
DEVFN float sigf(float x) { return 1.0f / (1.0f + __expf(-x)); }
DEVFN float tanhfast(float x) {
  float e = __expf(-2.0f * x);
  return (1.0f - e) / (1.0f + e);
}

// ---------------- main fused kernel ------------------------------------------------------
__global__ __launch_bounds__(256) void critic_kernel(const float* __restrict__ obs,
                                                     const float* __restrict__ act,
                                                     const unsigned char* __restrict__ ws,
                                                     float* __restrict__ out) {
  extern __shared__ unsigned char smem[];
  const int tid = threadIdx.x;
  const int lane = tid & 31;
  const int wave = tid >> 5;
  const int colL = lane & 15;
  const int rowBlk = blockIdx.x * 128;

  // Stage 0a: weights d_ws -> LDS via CDNA5 async global->LDS B128 (ASYNCcnt tracked,
  // no VGPR data movement; overlaps with the x-tile staging below).
  {
    const unsigned ldsbase = (unsigned)(uintptr_t)(void*)smem;
    for (int i = tid * 16; i < WBYTES; i += 256 * 16) {
      unsigned l = ldsbase + (unsigned)i;
      asm volatile("global_load_async_to_lds_b128 %0, %1, %2"
                   :
                   : "v"(l), "v"((unsigned)i), "s"(ws)
                   : "memory");
    }
  }
  // Stage 0b: x tile stage, vectorized: float4 global load -> 4 x f16 -> ds_store_b64.
  // Region boundaries (128 / 144 / 168) are all multiples of 4, so quads never straddle.
  {
    _Float16* xb0 = (_Float16*)(smem + OFF_XBUF);
    for (int i = tid; i < 128 * (XSTR / 4); i += 256) {
      int r = i / (XSTR / 4);
      int c0 = (i - r * (XSTR / 4)) * 4;
      size_t g = (size_t)(rowBlk + r);
      float4 v = make_float4(0.0f, 0.0f, 0.0f, 0.0f);
      if (c0 < S_)       v = *(const float4*)(obs + g * S_ + c0);
      else if (c0 < IN_) v = *(const float4*)(act + g * ADIM_ + (c0 - S_));
      v4h hv;
      hv[0] = (_Float16)v.x; hv[1] = (_Float16)v.y;
      hv[2] = (_Float16)v.z; hv[3] = (_Float16)v.w;
      *(v4h*)(xb0 + r * XSTR + c0) = hv;
    }
  }
  asm volatile("s_wait_asynccnt 0x0" ::: "memory");
  __syncthreads();

  const _Float16* wencT = (const _Float16*)(smem + OFF_WENCT);
  const _Float16* wobsT = (const _Float16*)(smem + OFF_WOBST);
  const _Float16* wiT   = (const _Float16*)(smem + OFF_WIT);
  const _Float16* whT   = (const _Float16*)(smem + OFF_WHT);
  const float* bencS = (const float*)(smem + OFF_BENC);
  const float* bobsS = (const float*)(smem + OFF_BOBS);
  const float* biS   = (const float*)(smem + OFF_BI);
  const float* bhS   = (const float*)(smem + OFF_BH);
  const float* wdecS = (const float*)(smem + OFF_WDEC);
  const float bdec   = ((const float*)(smem + OFF_BDEC))[0];

  _Float16* xb = (_Float16*)(smem + OFF_XBUF) + wave * 16 * XSTR;
  _Float16* hb = (_Float16*)(smem + OFF_HBUF) + wave * 16 * HSTR;
  _Float16* cb = (_Float16*)(smem + OFF_CBUF) + wave * 16 * HSTR;

  // ---- GEMM1: e = relu(x @ W_enc + b_enc), K = 160 (padded), store f16 -> hb
  {
    v8f acc[4] = {};
#pragma unroll
    for (int ks = 0; ks < 5; ++ks) {
      v16h a = ldsA(xb, XSTR, 32 * ks);
#pragma unroll
      for (int t = 0; t < 4; ++t) {
        v16h b = ldsB(wencT + 16 * t * XSTR, XSTR, 32 * ks);
        acc[t] = wmma_f(a, b, acc[t]);
      }
    }
#pragma unroll
    for (int t = 0; t < 4; ++t) {
      float be = bencS[16 * t + colL];
      v8f v = acc[t];
#pragma unroll
      for (int r = 0; r < 8; ++r) {
        float x = v[r] + be;
        v[r] = x > 0.0f ? x : 0.0f;
      }
      storeC16(hb, HSTR, t, v);
    }
  }
  __syncthreads();

  // ---- GEMM2: h = e @ W_obs + b_obs (kept in regs, C layout)
  v8f hcur[4];
  {
    v8f acc[4] = {};
#pragma unroll
    for (int ks = 0; ks < 2; ++ks) {
      v16h a = ldsA(hb, HSTR, 32 * ks);
#pragma unroll
      for (int t = 0; t < 4; ++t) {
        v16h b = ldsB(wobsT + 16 * t * HSTR, HSTR, 32 * ks);
        acc[t] = wmma_f(a, b, acc[t]);
      }
    }
#pragma unroll
    for (int t = 0; t < 4; ++t) {
      float bo = bobsS[16 * t + colL];
#pragma unroll
      for (int r = 0; r < 8; ++r) acc[t][r] += bo;
      hcur[t] = acc[t];
    }
  }
  // GRU1 needs h as A operand: overwrite hb (own-wave region; DS ops in order per wave)
#pragma unroll
  for (int t = 0; t < 4; ++t) storeC16(hb, HSTR, t, hcur[t]);
  __syncthreads();

  // ---- GRU cell 1 (c = 0): gi = bi, gh = h @ Wh + bh.  A fragments hoisted (loaded once).
  {
    v16h ah0 = ldsA(hb, HSTR, 0);
    v16h ah1 = ldsA(hb, HSTR, 32);
#pragma unroll
    for (int t = 0; t < 4; ++t) {
      v8f aR = {}, aZ = {}, aN = {};
#pragma unroll
      for (int ks = 0; ks < 2; ++ks) {
        v16h a = ks ? ah1 : ah0;
        v16h bR = ldsB(whT + (16 * t) * HSTR, HSTR, 32 * ks);
        aR = wmma_f(a, bR, aR);
        v16h bZ = ldsB(whT + (64 + 16 * t) * HSTR, HSTR, 32 * ks);
        aZ = wmma_f(a, bZ, aZ);
        v16h bN = ldsB(whT + (128 + 16 * t) * HSTR, HSTR, 32 * ks);
        aN = wmma_f(a, bN, aN);
      }
      const int c0 = 16 * t + colL;
      float bir = biS[c0], biz = biS[64 + c0], bin = biS[128 + c0];
      float bhr = bhS[c0], bhz = bhS[64 + c0], bhn = bhS[128 + c0];
#pragma unroll
      for (int r = 0; r < 8; ++r) {
        float rg = sigf(bir + aR[r] + bhr);
        float zg = sigf(biz + aZ[r] + bhz);
        float ng = tanhfast(bin + rg * (aN[r] + bhn));
        hcur[t][r] = (1.0f - zg) * ng + zg * hcur[t][r];
      }
    }
  }

  // ---- all-but-self agent mean: rows of a batch group are exactly the 8 acc VGPRs
  //      of one lane-half -> purely per-lane arithmetic. Stage h1 -> hb, c -> cb.
#pragma unroll
  for (int t = 0; t < 4; ++t) {
    v8f h1 = hcur[t];
    float s = h1[0] + h1[1] + h1[2] + h1[3] + h1[4] + h1[5] + h1[6] + h1[7];
    v8f cv;
#pragma unroll
    for (int r = 0; r < 8; ++r) cv[r] = (s - h1[r]) * 0.125f;
    storeC16(hb, HSTR, t, h1);
    storeC16(cb, HSTR, t, cv);
  }
  __syncthreads();

  // ---- GRU cell 2: gi = c @ Wi + bi, gh = h @ Wh + bh; decode dot W_dec fused in.
  //      A fragments (c and h1) hoisted out of the tile loop.
  float part[8] = {0, 0, 0, 0, 0, 0, 0, 0};
  {
    v16h ac0 = ldsA(cb, HSTR, 0);
    v16h ac1 = ldsA(cb, HSTR, 32);
    v16h ah0 = ldsA(hb, HSTR, 0);
    v16h ah1 = ldsA(hb, HSTR, 32);
#pragma unroll
    for (int t = 0; t < 4; ++t) {
      v8f iR = {}, iZ = {}, iN = {}, hR = {}, hZ = {}, hN = {};
#pragma unroll
      for (int ks = 0; ks < 2; ++ks) {
        v16h aC = ks ? ac1 : ac0;
        v16h aH = ks ? ah1 : ah0;
        v16h b;
        b = ldsB(wiT + (16 * t) * HSTR, HSTR, 32 * ks);        iR = wmma_f(aC, b, iR);
        b = ldsB(wiT + (64 + 16 * t) * HSTR, HSTR, 32 * ks);   iZ = wmma_f(aC, b, iZ);
        b = ldsB(wiT + (128 + 16 * t) * HSTR, HSTR, 32 * ks);  iN = wmma_f(aC, b, iN);
        b = ldsB(whT + (16 * t) * HSTR, HSTR, 32 * ks);        hR = wmma_f(aH, b, hR);
        b = ldsB(whT + (64 + 16 * t) * HSTR, HSTR, 32 * ks);   hZ = wmma_f(aH, b, hZ);
        b = ldsB(whT + (128 + 16 * t) * HSTR, HSTR, 32 * ks);  hN = wmma_f(aH, b, hN);
      }
      const int c0 = 16 * t + colL;
      float bir = biS[c0], biz = biS[64 + c0], bin = biS[128 + c0];
      float bhr = bhS[c0], bhz = bhS[64 + c0], bhn = bhS[128 + c0];
      float wd = wdecS[c0];
#pragma unroll
      for (int r = 0; r < 8; ++r) {
        float rg = sigf(iR[r] + bir + hR[r] + bhr);
        float zg = sigf(iZ[r] + biz + hZ[r] + bhz);
        float ng = tanhfast(iN[r] + bin + rg * (hN[r] + bhn));
        float hf = (1.0f - zg) * ng + zg * hcur[t][r];
        part[r] += hf * wd;
      }
    }
  }

  // ---- half-wave (16-lane) reduction of the decode dot, write [N,1] output
#pragma unroll
  for (int r = 0; r < 8; ++r) {
    float v = part[r];
    v += __shfl_xor(v, 1);
    v += __shfl_xor(v, 2);
    v += __shfl_xor(v, 4);
    v += __shfl_xor(v, 8);
    if (colL == 0) {
      int row = rowBlk + wave * 16 + r + ((lane & 16) ? 8 : 0);
      out[row] = v + bdec;
    }
  }
}

// ---------------- launch -----------------------------------------------------------------
extern "C" void kernel_launch(void* const* d_in, const int* in_sizes, int n_in,
                              void* d_out, int out_size, void* d_ws, size_t ws_size,
                              hipStream_t stream) {
  const float* obs  = (const float*)d_in[0];
  const float* act  = (const float*)d_in[1];
  const float* Wenc = (const float*)d_in[2];
  const float* benc = (const float*)d_in[3];
  const float* Wobs = (const float*)d_in[4];
  const float* bobs = (const float*)d_in[5];
  const float* Wi   = (const float*)d_in[6];
  const float* bi   = (const float*)d_in[7];
  const float* Wh   = (const float*)d_in[8];
  const float* bh   = (const float*)d_in[9];
  const float* Wdec = (const float*)d_in[10];
  const float* bdec = (const float*)d_in[11];
  unsigned char* ws = (unsigned char*)d_ws;
  float* out = (float*)d_out;

  prep_weights<<<64, 256, 0, stream>>>(Wenc, benc, Wobs, bobs, Wi, bi, Wh, bh, Wdec, bdec, ws);
  critic_kernel<<<NROWS / 128, 256, SMEM_BYTES, stream>>>(obs, act, ws, out);
}
